// Network_68375879352708
// MI455X (gfx1250) — matrix operand order repaired and checked by reading, not compiled, not run on hardware
//
#include <hip/hip_runtime.h>

// ---------------------------------------------------------------------------
// Types for CDNA5 WMMA (wave32): v_wmma_f32_16x16x32_f16
// ---------------------------------------------------------------------------
typedef __attribute__((ext_vector_type(16))) _Float16 v16h;
typedef __attribute__((ext_vector_type(8)))  _Float16 v8h;
typedef __attribute__((ext_vector_type(8)))  float    v8f;

union H16 { v16h v; v8h h[2]; };
union F8  { v8f  v; float f[8]; };

#define TM 128          // points per workgroup tile
#define HID 256         // hidden width

// LDS carve (bytes)
#define XA_BYTES   (TM * 64 * 2)
#define DD_BYTES   (TM * 64 * 2)
#define ACT_BYTES  (TM * HID * 2)
#define BW_OFF     (XA_BYTES + DD_BYTES + 2 * ACT_BYTES)   // 163840
#define BW_BYTES   (128 * 1024)                            // largest B panel
#define SHMEM_TOTAL (BW_OFF + BW_BYTES)                    // 294912 < 320KB

// ---------------------------------------------------------------------------
// Tensor Data Mover staging (gfx1250). This toolchain exposes the 6-arg
// builtin: (u32x4 g0, i32x8 g1, i32x4 g2, i32x4 g3, i32x8 extra, i32 cpol).
// Guarded: falls back to direct-global B reads if unavailable.
// ---------------------------------------------------------------------------
#if defined(__AMDGCN__) && __has_builtin(__builtin_amdgcn_tensor_load_to_lds) && \
    __has_builtin(__builtin_amdgcn_s_wait_tensorcnt)
#define USE_TDM 1
#else
#define USE_TDM 0
#endif

#if USE_TDM
typedef unsigned int v4u   __attribute__((ext_vector_type(4)));
typedef int          v4i_t __attribute__((ext_vector_type(4)));
typedef int          v8i_t __attribute__((ext_vector_type(8)));

// One-shot 1-D TDM copy: n_qwords * 8 bytes, global -> LDS byte offset lds_off.
// D# per CDNA5 ISA §8.3-8.5: group0 = {count=1, lds_addr, global_addr, type=2},
// group1 = {data_size=8B, tensor_dim0=n, tile_dim0=n, tile_dim1=1, stride=n}.
__device__ __forceinline__ void tdm_load_1d(unsigned lds_off, const void* gsrc,
                                            unsigned n_qwords)
{
  unsigned long long ga = (unsigned long long)(size_t)gsrc;
  v4u g0;
  g0[0] = 1u;                                       // count=1 (valid user D#)
  g0[1] = lds_off;                                  // lds_addr [63:32]
  g0[2] = (unsigned)(ga & 0xffffffffu);             // global_addr lo
  g0[3] = (unsigned)((ga >> 32) & 0x01ffffffu)      // global_addr hi (57-bit)
        | (2u << 30);                               // type=2 ("image")
  v8i_t g1;
  g1[0] = (int)(3u << 16);                          // wg_mask=0, data_size=3 (8B)
  g1[1] = (int)((n_qwords & 0xffffu) << 16);        // tensor_dim0[15:0]
  g1[2] = (int)(((n_qwords >> 16) & 0xffffu)        // tensor_dim0[31:16]
        | (1u << 16));                              // tensor_dim1 = 1
  g1[3] = (int)((n_qwords & 0xffffu) << 16);        // tile_dim0
  g1[4] = 1;                                        // tile_dim1=1, tile_dim2=0
  g1[5] = (int)n_qwords;                            // tensor_dim0_stride lo
  g1[6] = 0;                                        // stride hi / dim1_stride lo
  g1[7] = 0;
  v4i_t z4 = {0, 0, 0, 0};
  v8i_t z8 = {0, 0, 0, 0, 0, 0, 0, 0};
  __builtin_amdgcn_tensor_load_to_lds(g0, g1, z4, z4, z8, 0);
}
#endif

// Stage a packed-B panel for one layer. Returns the pointer GEMM should read.
__device__ __forceinline__ const _Float16*
stage_B(const _Float16* gsrc, unsigned bytes, _Float16* BW, int wave)
{
#if USE_TDM
  if (wave == 0) {                       // wave-uniform; TDM ignores EXEC
    tdm_load_1d(BW_OFF, gsrc, bytes >> 3);
    __builtin_amdgcn_s_wait_tensorcnt(0);
  }
  __syncthreads();                       // release other waves after DMA done
  return BW;
#else
  (void)bytes; (void)BW; (void)wave;
  return gsrc;
#endif
}

// ---------------------------------------------------------------------------
// Fragment loads.
// A fragment (16x32 f16): lane m (m=lane&15) holds row m; low half-wave holds
// K = {kb..kb+7, kb+16..kb+23}, high half-wave K+8. Two 16B LDS loads.
// B fragment (32x16 f16): packed so lane l's 16 f16 are contiguous (32B).
// ---------------------------------------------------------------------------
__device__ __forceinline__ v16h ld_a_frag(const _Float16* p) {
  H16 u;
  u.h[0] = *(const v8h*)(p);
  u.h[1] = *(const v8h*)(p + 16);
  return u.v;
}
__device__ __forceinline__ v16h ld_b_frag(const _Float16* p) {
  H16 u;
  u.h[0] = *(const v8h*)(p);
  u.h[1] = *(const v8h*)(p + 8);
  return u.v;
}

// ---------------------------------------------------------------------------
// Wave-level GEMM with compile-time epilogue (no branches in the hot path).
// OUT(16 x NT*16) = A(16 x KT*32) * B + bias [+relu] [+res] [+zr].
// ---------------------------------------------------------------------------
template<int KT, int NT, bool RELU, bool RES, bool ZR>
__device__ void gemm_layer(const _Float16* A, int aStride, int ktA,
                           const _Float16* A2,
                           const _Float16* Bpk,
                           const float* __restrict__ bias,
                           const _Float16* resBuf,
                           const float* __restrict__ zr,
                           _Float16* outBuf, int outStride,
                           int rowbase, int lane)
{
  const int ll = lane & 15;
  const int hi = lane >> 4;

  v16h af[KT];
#pragma unroll
  for (int kt = 0; kt < KT; ++kt) {
    if (kt < ktA)
      af[kt] = ld_a_frag(A  + (rowbase + ll) * aStride + kt * 32 + hi * 8);
    else
      af[kt] = ld_a_frag(A2 + (rowbase + ll) * 64 + (kt - ktA) * 32 + hi * 8);
  }

  for (int nt = 0; nt < NT; ++nt) {
    const int n0  = nt * 16;
    const int col = n0 + ll;
    if (nt + 1 < NT)
      __builtin_prefetch(Bpk + ((size_t)(nt + 1) * 32 + lane) * 16, 0, 1);

    const float b0 = bias[col];
    const float zv = ZR ? zr[col] : 0.0f;

    F8 acc;
#pragma unroll
    for (int v = 0; v < 8; ++v) acc.f[v] = b0;

#pragma unroll
    for (int kt = 0; kt < KT; ++kt) {
      v16h bf = ld_b_frag(Bpk + ((size_t)(kt * NT + nt) * 32 + lane) * 16);
      acc.v = __builtin_amdgcn_wmma_f32_16x16x32_f16(
          false, af[kt], false, bf, (short)0, acc.v, false, false);
    }

#pragma unroll
    for (int v = 0; v < 8; ++v) {
      float o = acc.f[v];
      if (RELU) o = fmaxf(o, 0.0f);
      const int row = rowbase + hi * 8 + v;
      if (RES) o += (float)resBuf[row * HID + col];
      if (ZR)  o += zv;
      outBuf[row * outStride + col] = (_Float16)o;
    }
  }
}

// ---------------------------------------------------------------------------
// Parameter bundle
// ---------------------------------------------------------------------------
struct NetParams {
  const float *inp, *view, *lc;
  float* out;
  const float *l0b;
  const float *zr;                 // 5 x 256, relu(pose@Wz^T+bz)
  const float *b1[5], *b2[5];
  const float *rgb1b, *alphaW, *alphaB, *rgb2W, *rgb2B;
  const _Float16 *pkL0, *pkRGB1;
  const _Float16 *pkW1[5], *pkW2[5];
  int N;
};

// ---------------------------------------------------------------------------
// Main fused kernel: PE -> l0 -> 5 residual blocks -> rgb1 -> heads
// ---------------------------------------------------------------------------
__global__ void __launch_bounds__(256)
nerf_main(NetParams P)
{
  extern __shared__ char smem[];
  _Float16* XA   = (_Float16*)smem;             // 128 x 64
  _Float16* DD   = XA   + TM * 64;              // 128 x 64
  _Float16* ACT0 = DD   + TM * 64;              // 128 x 256 (feat / res)
  _Float16* ACT1 = ACT0 + TM * HID;             // 128 x 256 (t / rgb hidden)
  _Float16* BW   = (_Float16*)(smem + BW_OFF);  // staged weight panel

  const int tid  = threadIdx.x;
  const int lane = tid & 31;
  const int wave = tid >> 5;
  const int rowbase = wave * 16;
  const int base = blockIdx.x * TM;

  // ---- Phase A: positional encodings ----
  if (tid < TM) {
    _Float16* xr = XA + tid * 64;
    _Float16* dr = DD + tid * 64;
    const int pidx = base + tid;
    if (pidx < P.N) {
      const float* ip = P.inp + (size_t)pidx * 4;
      float xi[4] = {ip[0], ip[1], ip[2], ip[3]};
#pragma unroll
      for (int c = 0; c < 4; ++c) xr[c] = (_Float16)xi[c];
      int o = 4;
#pragma unroll
      for (int l = 0; l < 6; ++l) {
        const float f = (float)(1 << l);
#pragma unroll
        for (int c = 0; c < 4; ++c) xr[o + c]     = (_Float16)__sinf(f * xi[c]);
#pragma unroll
        for (int c = 0; c < 4; ++c) xr[o + 4 + c] = (_Float16)__cosf(f * xi[c]);
        o += 8;
      }
      for (int c = 52; c < 64; ++c) xr[c] = (_Float16)0.0f;

      const float* vp  = P.view + (size_t)pidx * 3;
      const float* lcp = P.lc   + (size_t)pidx * 9;
      float dv[6];
      dv[0] = vp[0]; dv[1] = vp[1]; dv[2] = vp[2];
#pragma unroll
      for (int r = 0; r < 3; ++r)
        dv[3 + r] = lcp[r*3+0]*dv[0] + lcp[r*3+1]*dv[1] + lcp[r*3+2]*dv[2];
#pragma unroll
      for (int c = 0; c < 6; ++c) dr[c] = (_Float16)dv[c];
      o = 6;
#pragma unroll
      for (int l = 0; l < 4; ++l) {
        const float f = (float)(1 << l);
#pragma unroll
        for (int c = 0; c < 6; ++c) dr[o + c]     = (_Float16)__sinf(f * dv[c]);
#pragma unroll
        for (int c = 0; c < 6; ++c) dr[o + 6 + c] = (_Float16)__cosf(f * dv[c]);
        o += 12;
      }
      for (int c = 54; c < 64; ++c) dr[c] = (_Float16)0.0f;
    } else {
      for (int c = 0; c < 64; ++c) { xr[c] = (_Float16)0.0f; dr[c] = (_Float16)0.0f; }
    }
  }
  __syncthreads();

  // ---- l0: res0 = X @ W0^T + b0 + zr0 (fused) ----
  {
    const _Float16* B = stage_B(P.pkL0, 2 * 16 * 512 * 2, BW, wave);
    gemm_layer<2, 16, false, false, true>(XA, 64, 2, nullptr, B, P.l0b,
                                          nullptr, P.zr, ACT0, HID, rowbase, lane);
  }
  __syncthreads();

  // ---- 5 residual blocks ----
#pragma unroll 1
  for (int i = 0; i < 5; ++i) {
    {
      const _Float16* B = stage_B(P.pkW1[i], 8 * 16 * 512 * 2, BW, wave);
      gemm_layer<8, 16, true, false, false>(ACT0, HID, 8, nullptr, B, P.b1[i],
                                            nullptr, nullptr, ACT1, HID, rowbase, lane);
    }
    __syncthreads();
    {
      const _Float16* B = stage_B(P.pkW2[i], 8 * 16 * 512 * 2, BW, wave);
      if (i < 4)
        gemm_layer<8, 16, true, true, true>(ACT1, HID, 8, nullptr, B, P.b2[i],
                                            ACT0, P.zr + (i + 1) * HID,
                                            ACT0, HID, rowbase, lane);
      else
        gemm_layer<8, 16, true, true, false>(ACT1, HID, 8, nullptr, B, P.b2[i],
                                             ACT0, nullptr,
                                             ACT0, HID, rowbase, lane);
    }
    __syncthreads();
  }

  // ---- rgb1: relu([feat | d] @ W^T + b), K = 256 + 64(pad), NT = 8 ----
  {
    const _Float16* B = stage_B(P.pkRGB1, 10 * 8 * 512 * 2, BW, wave);
    gemm_layer<10, 8, true, false, false>(ACT0, HID, 8, DD, B, P.rgb1b,
                                          nullptr, nullptr, ACT1, HID, rowbase, lane);
  }
  __syncthreads();

  // ---- heads: alpha (1x256) and rgb2 (3x128), masked output ----
  if (tid < TM) {
    const int pidx = base + tid;
    if (pidx < P.N) {
      const float h = P.inp[(size_t)pidx * 4 + 3];
      const bool msk = (h < 0.15f) && (h > -0.5f);

      const _Float16* fr = ACT0 + tid * HID;
      float a = P.alphaB[0];
      for (int k = 0; k < HID; ++k) a += (float)fr[k] * P.alphaW[k];

      const _Float16* rr = ACT1 + tid * HID;
      float rgb[3];
#pragma unroll
      for (int c = 0; c < 3; ++c) {
        float s = P.rgb2B[c];
        for (int k = 0; k < 128; ++k) s += (float)rr[k] * P.rgb2W[c * 128 + k];
        rgb[c] = s;
      }
      float4 ov = msk ? make_float4(rgb[0], rgb[1], rgb[2], a)
                      : make_float4(0.f, 0.f, 0.f, 0.f);
      ((float4*)P.out)[pidx] = ov;
    }
  }
}

// ---------------------------------------------------------------------------
// Weight packer: W (out_c x in_c, f32 row-major) -> wave32 B-fragment layout.
// ---------------------------------------------------------------------------
__global__ void pack_weights(const float* __restrict__ W, _Float16* __restrict__ dst,
                             int NT, int in_c)
{
  const int frag = blockIdx.x;
  const int lane = threadIdx.x;
  const int kt = frag / NT;
  const int nt = frag % NT;
  const int n  = nt * 16 + (lane & 15);

  _Float16 tmp[16];
#pragma unroll
  for (int j = 0; j < 16; ++j) {
    const int k = kt * 32 + (lane >> 4) * 8 + (j & 7) + (j >> 3) * 16;
    const float w = (k < in_c) ? W[(size_t)n * in_c + k] : 0.0f;
    tmp[j] = (_Float16)w;
  }
  _Float16* dp = dst + ((size_t)frag * 32 + lane) * 16;
  *(v8h*)(dp)     = *(v8h*)(tmp);
  *(v8h*)(dp + 8) = *(v8h*)(tmp + 8);
}

// ---------------------------------------------------------------------------
// zr_i = relu(pose_embed @ Wz_i^T + bz_i)   (1x256; dispatch-constant)
// ---------------------------------------------------------------------------
__global__ void zr_kernel(const float* __restrict__ pose,
                          const float* __restrict__ Wz,
                          const float* __restrict__ bz,
                          float* __restrict__ out)
{
  const int n = threadIdx.x;
  float s = bz[n];
  for (int k = 0; k < 256; ++k) s += pose[k] * Wz[(size_t)n * 256 + k];
  out[n] = fmaxf(s, 0.0f);
}

// ---------------------------------------------------------------------------
// Launcher
// ---------------------------------------------------------------------------
extern "C" void kernel_launch(void* const* d_in, const int* in_sizes, int n_in,
                              void* d_out, int out_size, void* d_ws, size_t ws_size,
                              hipStream_t stream)
{
  (void)n_in; (void)out_size; (void)ws_size;

  const float* inp    = (const float*)d_in[0];
  const float* view   = (const float*)d_in[1];
  const float* lc     = (const float*)d_in[2];
  const float* pose   = (const float*)d_in[3];
  const float* alphaW = (const float*)d_in[4];
  const float* alphaB = (const float*)d_in[5];
  const float* l0W    = (const float*)d_in[6];
  const float* l0b    = (const float*)d_in[7];
  const float *Wz[5], *bz[5], *W1[5], *b1[5], *W2[5], *b2[5];
  for (int i = 0; i < 5; ++i) {
    const int b = 8 + 6 * i;
    Wz[i] = (const float*)d_in[b + 0];
    bz[i] = (const float*)d_in[b + 1];
    W1[i] = (const float*)d_in[b + 2];
    b1[i] = (const float*)d_in[b + 3];
    W2[i] = (const float*)d_in[b + 4];
    b2[i] = (const float*)d_in[b + 5];
  }
  const float* rgb1W = (const float*)d_in[38];
  const float* rgb1b = (const float*)d_in[39];
  const float* rgb2W = (const float*)d_in[40];
  const float* rgb2B = (const float*)d_in[41];

  const int N = in_sizes[0] / 4;

  // ---- workspace carve ----
  float* zr = (float*)d_ws;                               // 5*256 f32
  _Float16* pk = (_Float16*)((char*)d_ws + 8192);
  size_t off = 0;
  _Float16* pkL0 = pk + off;            off += (size_t)2  * 16 * 512;
  _Float16* pkW1[5]; _Float16* pkW2[5];
  for (int i = 0; i < 5; ++i) {
    pkW1[i] = pk + off;                 off += (size_t)8 * 16 * 512;
    pkW2[i] = pk + off;                 off += (size_t)8 * 16 * 512;
  }
  _Float16* pkRGB1 = pk + off;          off += (size_t)10 * 8 * 512;

  // ---- preprocessing kernels ----
  for (int i = 0; i < 5; ++i)
    zr_kernel<<<1, 256, 0, stream>>>(pose, Wz[i], bz[i], zr + i * 256);

  pack_weights<<<2 * 16, 32, 0, stream>>>(l0W, pkL0, 16, 52);
  for (int i = 0; i < 5; ++i) {
    pack_weights<<<8 * 16, 32, 0, stream>>>(W1[i], pkW1[i], 16, 256);
    pack_weights<<<8 * 16, 32, 0, stream>>>(W2[i], pkW2[i], 16, 256);
  }
  pack_weights<<<10 * 8, 32, 0, stream>>>(rgb1W, pkRGB1, 8, 310);

  // ---- main fused kernel ----
  NetParams P;
  P.inp = inp; P.view = view; P.lc = lc; P.out = (float*)d_out;
  P.l0b = l0b; P.zr = zr;
  for (int i = 0; i < 5; ++i) {
    P.b1[i] = b1[i]; P.b2[i] = b2[i];
    P.pkW1[i] = pkW1[i]; P.pkW2[i] = pkW2[i];
  }
  P.rgb1b = rgb1b; P.alphaW = alphaW; P.alphaB = alphaB;
  P.rgb2W = rgb2W; P.rgb2B = rgb2B;
  P.pkL0 = pkL0; P.pkRGB1 = pkRGB1;
  P.N = N;

  const int grid = (N + TM - 1) / TM;
  nerf_main<<<grid, 256, SHMEM_TOTAL, stream>>>(P);
}
